// Transformer_10514079941223
// MI455X (gfx1250) — matrix operand order repaired and checked by reading, not compiled, not run on hardware
//
#include <hip/hip_runtime.h>
#include <hip/hip_bf16.h>

// ---------------------------------------------------------------------------
// Model constants (must match reference)
// ---------------------------------------------------------------------------
constexpr int kB       = 4;
constexpr int kS       = 512;
constexpr int kTok     = kB * kS;      // 2048 tokens
constexpr int kD       = 512;          // d_model
constexpr int kFF      = 2048;         // d_ff
constexpr int kHeads   = 8;
constexpr int kDk      = 64;           // head dim
constexpr int kExperts = 8;
constexpr int kLayers  = 2;
constexpr int kOut     = 32000;
constexpr float kEps   = 1e-5f;

typedef __attribute__((ext_vector_type(16))) __bf16 v16bf;
typedef __attribute__((ext_vector_type(8)))  float  v8f;
typedef __attribute__((ext_vector_type(4)))  int    i32x4;

#define AS_GLOBAL __attribute__((address_space(1)))
#define AS_LOCAL  __attribute__((address_space(3)))

// CDNA5 async global->LDS path (ASYNCcnt-tracked), with graceful fallback.
#if defined(__has_builtin)
# if __has_builtin(__builtin_amdgcn_global_load_async_to_lds_b128) && \
     __has_builtin(__builtin_amdgcn_s_wait_asynccnt)
#  define USE_ASYNC_LDS 1
# endif
#endif

// ---------------------------------------------------------------------------
// WMMA fragment lane mappings (CDNA5 ISA 7.12.2, wave32), g = lane>>4, ln = lane&15
// A 16x32 bf16: row M=ln, element i: K = i + 16g (code's order, same as B)
//   -> one contiguous 16-float run at K = 16g
// B 32x16 bf16: col N=ln, element i: K = i + 16g (code's order, not ISA table) -> one contiguous 16-elem run
// C/D 16x16 f32: element r: M = r + 8g, N = ln
// ---------------------------------------------------------------------------

// Stage one 128x32 A tile (async DMA, fp32) and 32x128 B tile (transposed,
// converted to bf16) into the given LDS buffers.
__device__ __forceinline__ void stage_tile(
    const float* __restrict__ A, const float* __restrict__ W,
    int N, int K, int m0, int n0, int k0, int tid,
    float* lds_a, __bf16* lds_b)
{
#pragma unroll
  for (int r = 0; r < 4; ++r) {
    const int chunk = tid + r * 256;           // 0..1023
    const int row   = chunk >> 3;              // 8 x 16B chunks per 128B row
    const int c4    = (chunk & 7) << 2;
    const float* gsrc = A + (size_t)(m0 + row) * K + k0 + c4;
    float* ldst = lds_a + row * 32 + c4;
#ifdef USE_ASYNC_LDS
    __builtin_amdgcn_global_load_async_to_lds_b128(
        (AS_GLOBAL i32x4*)gsrc, (AS_LOCAL i32x4*)ldst, 0, 0);
#else
    *(float4*)ldst = *(const float4*)gsrc;
#endif
  }
#pragma unroll
  for (int r = 0; r < 4; ++r) {
    const int chunk = tid + r * 256;           // 0..1023, 4 floats each
    const int kr    = chunk >> 5;
    const int nc    = (chunk & 31) << 2;
    const float4 v = *(const float4*)&W[(size_t)(k0 + kr) * N + n0 + nc];
    lds_b[(nc + 0) * 32 + kr] = (__bf16)v.x;
    lds_b[(nc + 1) * 32 + kr] = (__bf16)v.y;
    lds_b[(nc + 2) * 32 + kr] = (__bf16)v.z;
    lds_b[(nc + 3) * 32 + kr] = (__bf16)v.w;
  }
}

// Tiled GEMM: C[M,N] = epilogue(A[M,K] @ W[K,N] + bias)
// 256 threads: 8 waves in 4(M) x 2(N) layout; macro tile 128x128; K step 32;
// double-buffered LDS pipeline (stage tile t+1 while computing tile t).
// MODE: 0 = bias; 1 = bias+gelu(erf); 2 = bias+silu; 3 = bias, *cw, +=C
template <int MODE>
__global__ __launch_bounds__(256) void gemm_wmma(
    const float* __restrict__ A, const float* __restrict__ W,
    const float* __restrict__ bias, float* __restrict__ C,
    int M, int N, int K,
    const float* __restrict__ cw, int expIdx)
{
  __shared__ __align__(16) float  lds_a[2][128 * 32];  // 2 x 16 KB
  __shared__ __align__(32) __bf16 lds_b[2][128 * 32];  // 2 x 8 KB

  const int tid  = threadIdx.x;
  const int lane = tid & 31;
  const int g    = lane >> 4;
  const int ln   = lane & 15;
  const int wave = tid >> 5;
  const int wm   = (wave >> 1) * 32;   // wave row offset within macro tile
  const int wn   = (wave & 1) * 64;    // wave col offset within macro tile
  const int n0   = blockIdx.x * 128;
  const int m0   = blockIdx.y * 128;

  v8f acc[2][4];
#pragma unroll
  for (int tm = 0; tm < 2; ++tm)
#pragma unroll
    for (int tn = 0; tn < 4; ++tn)
#pragma unroll
      for (int r = 0; r < 8; ++r) acc[tm][tn][r] = 0.f;

  // prologue: stage tile 0
  stage_tile(A, W, N, K, m0, n0, 0, tid, lds_a[0], lds_b[0]);
#ifdef USE_ASYNC_LDS
  __builtin_amdgcn_s_wait_asynccnt(0);
#endif
  __syncthreads();

  int cur = 0;
  for (int k0 = 0; k0 < K; k0 += 32) {
    const bool more = (k0 + 32) < K;
    if (more)
      stage_tile(A, W, N, K, m0, n0, k0 + 32, tid, lds_a[cur ^ 1], lds_b[cur ^ 1]);

    // ---- A fragments (2 per wave): contiguous LDS reads + cvt to bf16
    v16bf afr[2];
#pragma unroll
    for (int tm = 0; tm < 2; ++tm) {
      const int row = wm + tm * 16 + ln;
      const float* ar = &lds_a[cur][row * 32 + 16 * g];
#pragma unroll
      for (int i = 0; i < 8; ++i) afr[tm][i]     = (__bf16)ar[i];
#pragma unroll
      for (int i = 0; i < 8; ++i) afr[tm][i + 8] = (__bf16)ar[8 + i];
    }
    // ---- B fragments + 8 WMMAs
#pragma unroll
    for (int tn = 0; tn < 4; ++tn) {
      const int col = wn + tn * 16 + ln;
      const v16bf bfr = *(const v16bf*)&lds_b[cur][col * 32 + 16 * g];
      acc[0][tn] = __builtin_amdgcn_wmma_f32_16x16x32_bf16(
          false, afr[0], false, bfr, (short)0, acc[0][tn], false, false);
      acc[1][tn] = __builtin_amdgcn_wmma_f32_16x16x32_bf16(
          false, afr[1], false, bfr, (short)0, acc[1][tn], false, false);
    }

    if (more) {
#ifdef USE_ASYNC_LDS
      __builtin_amdgcn_s_wait_asynccnt(0);
#endif
      __syncthreads();
    }
    cur ^= 1;
  }

  // ---- epilogue (specialized per MODE)
#pragma unroll
  for (int tn = 0; tn < 4; ++tn) {
    const int col = n0 + wn + tn * 16 + ln;
    const float bs = bias ? bias[col] : 0.f;
#pragma unroll
    for (int tm = 0; tm < 2; ++tm) {
#pragma unroll
      for (int r = 0; r < 8; ++r) {
        const int row = m0 + wm + tm * 16 + r + 8 * g;
        float v = acc[tm][tn][r] + bs;
        if constexpr (MODE == 1) {
          v = 0.5f * v * (1.f + erff(v * 0.70710678118654752f));
        } else if constexpr (MODE == 2) {
          v = v * __fdividef(1.f, 1.f + __expf(-v));
        }
        if constexpr (MODE == 3) {
          const float wgt = cw[row * kExperts + expIdx];
          C[(size_t)row * N + col] += wgt * v;
        } else {
          C[(size_t)row * N + col] = v;
        }
      }
    }
  }
}

// ---------------------------------------------------------------------------
// Embedding gather * sqrt(D) + RoPE (pairs over full d_model, as reference)
// ---------------------------------------------------------------------------
__global__ __launch_bounds__(256) void embed_rope(
    const int* __restrict__ toks, const float* __restrict__ emb,
    float* __restrict__ out)
{
  const int tok = blockIdx.x;
  const int i   = threadIdx.x;           // pair index 0..255
  const int s   = tok % kS;
  const int id  = toks[tok];
  const float scale = 22.62741699796952f; // sqrt(512)
  const float x1 = emb[(size_t)id * kD + 2 * i]     * scale;
  const float x2 = emb[(size_t)id * kD + 2 * i + 1] * scale;
  const float freq = powf(10000.f, -((float)(2 * i)) / (float)kD);
  const float ang  = (float)s * freq;
  const float sn = sinf(ang), cs = cosf(ang);
  out[(size_t)tok * kD + 2 * i]     = x1 * cs - x2 * sn;
  out[(size_t)tok * kD + 2 * i + 1] = x1 * sn + x2 * cs;
}

// ---------------------------------------------------------------------------
// Attention core: softmax(QK^T/sqrt(dk)) V for one (b, h, q) per block.
// ---------------------------------------------------------------------------
__global__ __launch_bounds__(256) void attn_kernel(
    const float* __restrict__ Q, const float* __restrict__ K,
    const float* __restrict__ V, float* __restrict__ O)
{
  __shared__ float sc[kS];
  __shared__ float red[256];
  const int q = blockIdx.x;
  const int h = blockIdx.y;
  const int b = blockIdx.z;
  const int t = threadIdx.x;
  const float scale = 0.125f;            // 1/sqrt(64)

  const float* qp = Q + ((size_t)(b * kS + q) * kD) + h * kDk;
  for (int k = t; k < kS; k += 256) {
    const float* kp = K + ((size_t)(b * kS + k) * kD) + h * kDk;
    float s = 0.f;
#pragma unroll
    for (int d = 0; d < kDk; ++d) s += qp[d] * kp[d];
    sc[k] = s * scale;
  }
  __syncthreads();

  float lm = -1e30f;
  for (int k = t; k < kS; k += 256) lm = fmaxf(lm, sc[k]);
  red[t] = lm;
  __syncthreads();
  for (int s = 128; s > 0; s >>= 1) {
    if (t < s) red[t] = fmaxf(red[t], red[t + s]);
    __syncthreads();
  }
  const float mx = red[0];
  __syncthreads();

  float ls = 0.f;
  for (int k = t; k < kS; k += 256) {
    const float e = __expf(sc[k] - mx);
    sc[k] = e;
    ls += e;
  }
  red[t] = ls;
  __syncthreads();
  for (int s = 128; s > 0; s >>= 1) {
    if (t < s) red[t] += red[t + s];
    __syncthreads();
  }
  const float inv = 1.f / red[0];
  __syncthreads();

  if (t < kDk) {
    float o = 0.f;
    for (int k = 0; k < kS; ++k)
      o += sc[k] * V[((size_t)(b * kS + k) * kD) + h * kDk + t];
    O[((size_t)(b * kS + q) * kD) + h * kDk + t] = o * inv;
  }
}

// ---------------------------------------------------------------------------
// In-place LayerNorm of (x + add): x = LN(x + add) * g + b. One row / block.
// ---------------------------------------------------------------------------
__global__ __launch_bounds__(256) void ln_residual(
    float* __restrict__ x, const float* __restrict__ add,
    const float* __restrict__ gw, const float* __restrict__ bw)
{
  __shared__ float red[256];
  const int row = blockIdx.x;
  const int t   = threadIdx.x;
  const size_t base = (size_t)row * kD;
  const float v0 = x[base + t]       + add[base + t];
  const float v1 = x[base + t + 256] + add[base + t + 256];
  red[t] = v0 + v1;
  __syncthreads();
  for (int s = 128; s > 0; s >>= 1) {
    if (t < s) red[t] += red[t + s];
    __syncthreads();
  }
  const float mean = red[0] * (1.f / (float)kD);
  __syncthreads();
  const float d0 = v0 - mean, d1 = v1 - mean;
  red[t] = d0 * d0 + d1 * d1;
  __syncthreads();
  for (int s = 128; s > 0; s >>= 1) {
    if (t < s) red[t] += red[t + s];
    __syncthreads();
  }
  const float rstd = rsqrtf(red[0] * (1.f / (float)kD) + kEps);
  x[base + t]       = d0 * rstd * gw[t]       + bw[t];
  x[base + t + 256] = d1 * rstd * gw[t + 256] + bw[t + 256];
}

// ---------------------------------------------------------------------------
// MoE gating: per token, gates = x@gw+gb; top-2 softmax -> dense cw[tok][8]
// ---------------------------------------------------------------------------
__global__ __launch_bounds__(256) void gate_topk(
    const float* __restrict__ x, const float* __restrict__ gw,
    const float* __restrict__ gb, float* __restrict__ cw)
{
  const int tok = blockIdx.x * blockDim.x + threadIdx.x;
  if (tok >= kTok) return;
  float gt[kExperts];
#pragma unroll
  for (int e = 0; e < kExperts; ++e) gt[e] = gb[e];
  const size_t base = (size_t)tok * kD;
  for (int d = 0; d < kD; ++d) {
    const float xv = x[base + d];
#pragma unroll
    for (int e = 0; e < kExperts; ++e) gt[e] += xv * gw[d * kExperts + e];
  }
  int i1 = 0;
#pragma unroll
  for (int e = 1; e < kExperts; ++e) if (gt[e] > gt[i1]) i1 = e;
  int i2 = (i1 == 0) ? 1 : 0;
#pragma unroll
  for (int e = 0; e < kExperts; ++e)
    if (e != i1 && gt[e] > gt[i2]) i2 = e;
  const float e2 = __expf(gt[i2] - gt[i1]);
  const float w1 = 1.f / (1.f + e2);
  const float w2 = e2 / (1.f + e2);
#pragma unroll
  for (int e = 0; e < kExperts; ++e) cw[(size_t)tok * kExperts + e] = 0.f;
  cw[(size_t)tok * kExperts + i1] = w1;
  cw[(size_t)tok * kExperts + i2] = w2;
}

// ---------------------------------------------------------------------------
// Host orchestration
// ---------------------------------------------------------------------------
extern "C" void kernel_launch(void* const* d_in, const int* in_sizes, int n_in,
                              void* d_out, int out_size, void* d_ws, size_t ws_size,
                              hipStream_t stream) {
  (void)in_sizes; (void)n_in; (void)out_size; (void)ws_size;

  int idx = 0;
  const float* emb    = (const float*)d_in[idx++];
  const float* out_w  = (const float*)d_in[idx++];
  const float* out_b  = (const float*)d_in[idx++];
  const float* e_q_w = (const float*)d_in[idx++]; const float* e_q_b = (const float*)d_in[idx++];
  const float* e_k_w = (const float*)d_in[idx++]; const float* e_k_b = (const float*)d_in[idx++];
  const float* e_v_w = (const float*)d_in[idx++]; const float* e_v_b = (const float*)d_in[idx++];
  const float* e_o_w = (const float*)d_in[idx++]; const float* e_o_b = (const float*)d_in[idx++];
  const float* e_n1_g = (const float*)d_in[idx++]; const float* e_n1_b = (const float*)d_in[idx++];
  const float* e_n3_g = (const float*)d_in[idx++]; const float* e_n3_b = (const float*)d_in[idx++];
  const float* e_gate_w = (const float*)d_in[idx++]; const float* e_gate_b = (const float*)d_in[idx++];
  const float* e_w1 = (const float*)d_in[idx++]; const float* e_b1 = (const float*)d_in[idx++];
  const float* e_w2 = (const float*)d_in[idx++]; const float* e_b2 = (const float*)d_in[idx++];
  const float* d_q_w = (const float*)d_in[idx++]; const float* d_q_b = (const float*)d_in[idx++];
  const float* d_k_w = (const float*)d_in[idx++]; const float* d_k_b = (const float*)d_in[idx++];
  const float* d_v_w = (const float*)d_in[idx++]; const float* d_v_b = (const float*)d_in[idx++];
  const float* d_o_w = (const float*)d_in[idx++]; const float* d_o_b = (const float*)d_in[idx++];
  const float* d_cq_w = (const float*)d_in[idx++]; const float* d_cq_b = (const float*)d_in[idx++];
  const float* d_ck_w = (const float*)d_in[idx++]; const float* d_ck_b = (const float*)d_in[idx++];
  const float* d_cv_w = (const float*)d_in[idx++]; const float* d_cv_b = (const float*)d_in[idx++];
  const float* d_co_w = (const float*)d_in[idx++]; const float* d_co_b = (const float*)d_in[idx++];
  const float* d_n1_g = (const float*)d_in[idx++]; const float* d_n1_b = (const float*)d_in[idx++];
  const float* d_n3_g = (const float*)d_in[idx++]; const float* d_n3_b = (const float*)d_in[idx++];
  const float* d_n2_g = (const float*)d_in[idx++]; const float* d_n2_b = (const float*)d_in[idx++];
  const float* d_gate_w = (const float*)d_in[idx++]; const float* d_gate_b = (const float*)d_in[idx++];
  const float* d_w1 = (const float*)d_in[idx++]; const float* d_b1 = (const float*)d_in[idx++];
  const float* d_w2 = (const float*)d_in[idx++]; const float* d_b2 = (const float*)d_in[idx++];
  const int* src = (const int*)d_in[idx++];
  const int* tgt = (const int*)d_in[idx++];

  float* wsf = (float*)d_ws;
  size_t off = 0;
  auto alloc = [&](size_t n) { float* p = wsf + off; off += (n + 63) & ~(size_t)63; return p; };
  float* xe      = alloc((size_t)kTok * kD);
  float* xd      = alloc((size_t)kTok * kD);
  float* qb      = alloc((size_t)kTok * kD);
  float* kb      = alloc((size_t)kTok * kD);
  float* vb      = alloc((size_t)kTok * kD);
  float* ctx     = alloc((size_t)kTok * kD);
  float* proj    = alloc((size_t)kTok * kD);
  float* moe_acc = alloc((size_t)kTok * kD);
  float* hbuf    = alloc((size_t)kTok * kFF);
  float* cwb     = alloc((size_t)kTok * kExperts);

  auto gemm = [&](const float* A, const float* Wm, const float* bias, float* Cm,
                  int M, int N, int Kd, int mode, const float* cwp, int e) {
    dim3 grid(N / 128, M / 128);
    switch (mode) {
      case 0: gemm_wmma<0><<<grid, 256, 0, stream>>>(A, Wm, bias, Cm, M, N, Kd, cwp, e); break;
      case 1: gemm_wmma<1><<<grid, 256, 0, stream>>>(A, Wm, bias, Cm, M, N, Kd, cwp, e); break;
      case 2: gemm_wmma<2><<<grid, 256, 0, stream>>>(A, Wm, bias, Cm, M, N, Kd, cwp, e); break;
      default: gemm_wmma<3><<<grid, 256, 0, stream>>>(A, Wm, bias, Cm, M, N, Kd, cwp, e); break;
    }
  };

  auto mha = [&](const float* qsrc, const float* kvsrc,
                 const float* wq, const float* bq, const float* wk, const float* bk,
                 const float* wv, const float* bv, const float* wo, const float* bo,
                 float* outp) {
    gemm(qsrc,  wq, bq, qb, kTok, kD, kD, 0, nullptr, 0);
    gemm(kvsrc, wk, bk, kb, kTok, kD, kD, 0, nullptr, 0);
    gemm(kvsrc, wv, bv, vb, kTok, kD, kD, 0, nullptr, 0);
    attn_kernel<<<dim3(kS, kHeads, kB), 256, 0, stream>>>(qb, kb, vb, ctx);
    gemm(ctx, wo, bo, outp, kTok, kD, kD, 0, nullptr, 0);
  };

  auto ln = [&](float* x, const float* add, const float* g, const float* b) {
    ln_residual<<<kTok, 256, 0, stream>>>(x, add, g, b);
  };

  auto moe = [&](float* x, const float* gw, const float* gb,
                 const float* w1, const float* b1, const float* w2, const float* b2) {
    gate_topk<<<kTok / 256, 256, 0, stream>>>(x, gw, gb, cwb);
    (void)hipMemsetAsync(moe_acc, 0, (size_t)kTok * kD * sizeof(float), stream);
    for (int e = 0; e < kExperts; ++e) {
      gemm(x, w1 + (size_t)e * kD * kFF, b1 + (size_t)e * kFF, hbuf,
           kTok, kFF, kD, (e % 2 == 0) ? 1 : 2, nullptr, 0);
      gemm(hbuf, w2 + (size_t)e * kFF * kD, b2 + (size_t)e * kD, moe_acc,
           kTok, kD, kFF, 3, cwb, e);
    }
  };

  const size_t DD = (size_t)kD * kD;

  // ---------------- encoder ----------------
  embed_rope<<<kTok, 256, 0, stream>>>(src, emb, xe);
  for (int l = 0; l < kLayers; ++l) {
    mha(xe, xe,
        e_q_w + l * DD, e_q_b + l * kD, e_k_w + l * DD, e_k_b + l * kD,
        e_v_w + l * DD, e_v_b + l * kD, e_o_w + l * DD, e_o_b + l * kD, proj);
    ln(xe, proj, e_n1_g + l * kD, e_n1_b + l * kD);
    moe(xe, e_gate_w + (size_t)l * kD * kExperts, e_gate_b + l * kExperts,
        e_w1 + (size_t)l * kExperts * kD * kFF, e_b1 + (size_t)l * kExperts * kFF,
        e_w2 + (size_t)l * kExperts * kFF * kD, e_b2 + (size_t)l * kExperts * kD);
    ln(xe, moe_acc, e_n3_g + l * kD, e_n3_b + l * kD);
  }

  // ---------------- decoder ----------------
  embed_rope<<<kTok, 256, 0, stream>>>(tgt, emb, xd);
  for (int l = 0; l < kLayers; ++l) {
    mha(xd, xd,
        d_q_w + l * DD, d_q_b + l * kD, d_k_w + l * DD, d_k_b + l * kD,
        d_v_w + l * DD, d_v_b + l * kD, d_o_w + l * DD, d_o_b + l * kD, proj);
    ln(xd, proj, d_n1_g + l * kD, d_n1_b + l * kD);
    mha(xd, xe,
        d_cq_w + l * DD, d_cq_b + l * kD, d_ck_w + l * DD, d_ck_b + l * kD,
        d_cv_w + l * DD, d_cv_b + l * kD, d_co_w + l * DD, d_co_b + l * kD, proj);
    ln(xd, proj, d_n2_g + l * kD, d_n2_b + l * kD);
    moe(xd, d_gate_w + (size_t)l * kD * kExperts, d_gate_b + l * kExperts,
        d_w1 + (size_t)l * kExperts * kD * kFF, d_b1 + (size_t)l * kExperts * kFF,
        d_w2 + (size_t)l * kExperts * kFF * kD, d_b2 + (size_t)l * kExperts * kD);
    ln(xd, moe_acc, d_n3_g + l * kD, d_n3_b + l * kD);
  }

  // ---------------- output projection ----------------
  gemm(xd, out_w, out_b, (float*)d_out, kTok, kOut, kD, 0, nullptr, 0);
}
